// MultimodalFusion_45775761440875
// MI455X (gfx1250) — matrix operand order repaired and checked by reading, not compiled, not run on hardware
//
#include <hip/hip_runtime.h>

// ---------------------------------------------------------------------------
// MultimodalFusion forward for MI455X (gfx1250, wave32, WMMA bf16 16x16x32).
// All matmuls run through v_wmma_f32_16x16x32_bf16 with f32 accumulation.
// Weight tiles (already bf16 in memory) are staged to LDS with the CDNA5
// async copy path (global_load_async_to_lds_b128 + s_wait_asynccnt).
// ---------------------------------------------------------------------------

typedef __attribute__((ext_vector_type(16))) __bf16 v16bf;
typedef __attribute__((ext_vector_type(8)))  float  v8f;

union BF16x16 { v16bf v; unsigned short u[16]; };

__device__ __forceinline__ unsigned short f2bfbits(float f) {
  unsigned u = __float_as_uint(f);
  unsigned r = u + 0x7FFFu + ((u >> 16) & 1u);   // round-to-nearest-even
  return (unsigned short)(r >> 16);
}

// ---------------- WMMA fragment builders (A/B in contiguous k order, k = 16*hi + s) --

// A fragment 16x32 bf16 from LDS tile laid out [row][32] (row-major, stride 32).
__device__ __forceinline__ v16bf frag_a_lds(const unsigned short* tile, int row0, int lane) {
  BF16x16 f;
  int m = lane & 15, hi = lane >> 4;
  const unsigned short* r = tile + (row0 + m) * 32;
#pragma unroll
  for (int v = 0; v < 8; ++v) {
    int k = (v << 1) + (hi << 4);                         // 2v + 16*hi
    f.u[2 * v]     = r[k];
    f.u[2 * v + 1] = r[k + 1];
  }
  return f.v;
}

// B fragment 32x16 bf16, B(k,n) from LDS tile stored transposed: [n][k] stride 32.
__device__ __forceinline__ v16bf frag_b_lds_t(const unsigned short* tile, int col0, int lane) {
  BF16x16 f;
  int n = lane & 15, hi = lane >> 4;
  const unsigned short* r = tile + (col0 + n) * 32;
#pragma unroll
  for (int v = 0; v < 8; ++v) {
    int k = (v << 1) + (hi << 4);                         // 2v + 16*hi
    f.u[2 * v]     = r[k];
    f.u[2 * v + 1] = r[k + 1];
  }
  return f.v;
}

// B fragment 32x16 bf16, B(k,n) from LDS tile stored natural: [k][n] stride 64.
__device__ __forceinline__ v16bf frag_b_lds_n(const unsigned short* tile, int col0, int lane) {
  BF16x16 f;
  int n = lane & 15, hi = lane >> 4;
#pragma unroll
  for (int v = 0; v < 8; ++v) {
    int k = (v << 1) + (hi << 4);
    f.u[2 * v]     = tile[k * 64 + col0 + n];
    f.u[2 * v + 1] = tile[(k + 1) * 64 + col0 + n];
  }
  return f.v;
}

// C/D 16x16 f32 store (lane 0-15: M=v, lane 16-31: M=v+8; N=lane&15).
__device__ __forceinline__ void store_ctile(float* C, long long ldc, int row0, int col0,
                                            int lane, v8f acc, const float* bias, int mode) {
  int n = lane & 15, hi = lane >> 4;
#pragma unroll
  for (int v = 0; v < 8; ++v) {
    int m = v + (hi << 3);
    float x = acc[v];
    if (bias) x += bias[col0 + n];
    if (mode == 1) x = (fabsf(x) > 0.01f) ? x : 0.0f;    // selective routing
    C[(long long)(row0 + m) * ldc + (col0 + n)] = x;
  }
}

// ---------------- GEMM: C[M,N] = A_f32[M,K] @ Wbf16[N,K]^T + bias ------------
// 64x64 tile per 256-thread block (8 waves; each wave two 16x16 WMMA tiles).
// B tile is staged by the CDNA5 async global->LDS DMA path (ASYNCcnt), the
// A tile goes through VGPRs for the f32->bf16 conversion, overlapping both.
__global__ __launch_bounds__(256) void gemm_bias_kernel(
    const float* __restrict__ A, int lda,
    const unsigned short* __restrict__ Wbf,   // [N,K] bf16 bits
    const float* __restrict__ bias,
    float* __restrict__ C, int ldc,
    int K, int mode) {
  __shared__ unsigned short As[64 * 32];
  __shared__ unsigned short Bs[64 * 32];
  int tid = threadIdx.x, lane = tid & 31, wv = tid >> 5;
  int wr = wv & 3, wc = wv >> 2;                         // wr 0..3, wc 0..1
  int m0 = blockIdx.y << 6, n0 = blockIdx.x << 6;
  int lr = tid >> 2, lc = (tid & 3) << 3;
  // Per-thread LDS byte address for the async B128 copy: thread t owns
  // Bs[lr*32+lc .. +7] which is byte offset t*16 from &Bs[0] (contiguous).
  unsigned bs_lds = (unsigned)(unsigned long long)(&Bs[0]) + (unsigned)(tid * 16);
  v8f acc0 = {}, acc1 = {};
  for (int k0 = 0; k0 < K; k0 += 32) {
    __syncthreads();
    // 1) kick off async DMA of the bf16 weight tile straight into LDS
    const unsigned short* bsrc = Wbf + (long long)(n0 + lr) * K + k0 + lc;
    asm volatile("global_load_async_to_lds_b128 %0, %1, off"
                 :: "v"(bs_lds), "v"(bsrc)
                 : "memory");
    // 2) meanwhile convert the f32 A tile to bf16 through VGPRs
    const float* as = A + (long long)(m0 + lr) * lda + k0 + lc;
#pragma unroll
    for (int j = 0; j < 8; ++j) As[lr * 32 + lc + j] = f2bfbits(as[j]);
    if (k0 + 32 < K) __builtin_prefetch(as + 32, 0, 0);  // global_prefetch_b8
    // 3) fence the async copy, then the workgroup barrier
    asm volatile("s_wait_asynccnt 0" ::: "memory");
    __syncthreads();
    v16bf af = frag_a_lds(As, wr * 16, lane);
    v16bf b0 = frag_b_lds_t(Bs, wc * 32, lane);
    v16bf b1 = frag_b_lds_t(Bs, wc * 32 + 16, lane);
    acc0 = __builtin_amdgcn_wmma_f32_16x16x32_bf16(false, af, false, b0, (short)0, acc0, false, false);
    acc1 = __builtin_amdgcn_wmma_f32_16x16x32_bf16(false, af, false, b1, (short)0, acc1, false, false);
  }
  store_ctile(C, ldc, m0 + wr * 16, n0 + wc * 32,      lane, acc0, bias, mode);
  store_ctile(C, ldc, m0 + wr * 16, n0 + wc * 32 + 16, lane, acc1, bias, mode);
}

// ---------------- Attention scores: S[h] = Q_h @ K_h^T (one batch) -----------
// Q,K: [512,1024] f32 (head h at column h*128). Sc: [8][512][512].
__global__ __launch_bounds__(256) void attn_scores_kernel(
    const float* __restrict__ Q, const float* __restrict__ Km, float* __restrict__ Sc) {
  __shared__ unsigned short As[64 * 32];
  __shared__ unsigned short Bs[64 * 32];
  int tid = threadIdx.x, lane = tid & 31, wv = tid >> 5;
  int wr = wv & 3, wc = wv >> 2;
  int m0 = blockIdx.y << 6, n0 = blockIdx.x << 6;
  int h = blockIdx.z;
  const float* Qh = Q + h * 128;
  const float* Kh = Km + h * 128;
  float* Sh = Sc + (long long)h * 512 * 512;
  int lr = tid >> 2, lc = (tid & 3) << 3;
  v8f acc0 = {}, acc1 = {};
  for (int k0 = 0; k0 < 128; k0 += 32) {
    __syncthreads();
    const float* as = Qh + (long long)(m0 + lr) * 1024 + k0 + lc;
#pragma unroll
    for (int j = 0; j < 8; ++j) As[lr * 32 + lc + j] = f2bfbits(as[j]);
    const float* bs = Kh + (long long)(n0 + lr) * 1024 + k0 + lc;
#pragma unroll
    for (int j = 0; j < 8; ++j) Bs[lr * 32 + lc + j] = f2bfbits(bs[j]);
    __syncthreads();
    v16bf af = frag_a_lds(As, wr * 16, lane);
    v16bf b0 = frag_b_lds_t(Bs, wc * 32, lane);
    v16bf b1 = frag_b_lds_t(Bs, wc * 32 + 16, lane);
    acc0 = __builtin_amdgcn_wmma_f32_16x16x32_bf16(false, af, false, b0, (short)0, acc0, false, false);
    acc1 = __builtin_amdgcn_wmma_f32_16x16x32_bf16(false, af, false, b1, (short)0, acc1, false, false);
  }
  store_ctile(Sh, 512, m0 + wr * 16, n0 + wc * 32,      lane, acc0, nullptr, 0);
  store_ctile(Sh, 512, m0 + wr * 16, n0 + wc * 32 + 16, lane, acc1, nullptr, 0);
}

// ---------------- Attention output: O_h = P_h @ V_h (one batch) --------------
// P: [8][512][512] probs; V,O: [512,1024] (head h at column h*128). N=128.
__global__ __launch_bounds__(256) void attn_out_kernel(
    const float* __restrict__ P, const float* __restrict__ V, float* __restrict__ O) {
  __shared__ unsigned short As[64 * 32];
  __shared__ unsigned short Bs[32 * 64];
  int tid = threadIdx.x, lane = tid & 31, wv = tid >> 5;
  int wr = wv & 3, wc = wv >> 2;
  int m0 = blockIdx.y << 6, n0 = blockIdx.x << 6;     // n0 in {0,64} within head
  int h = blockIdx.z;
  const float* Ph = P + (long long)h * 512 * 512;
  const float* Vh = V + h * 128;
  float* Oh = O + h * 128;
  int lr = tid >> 2, lc = (tid & 3) << 3;             // A tile 64x32
  int kr = tid >> 3, kc = (tid & 7) << 3;             // B tile 32x64
  v8f acc0 = {}, acc1 = {};
  for (int k0 = 0; k0 < 512; k0 += 32) {
    __syncthreads();
    const float* as = Ph + (long long)(m0 + lr) * 512 + k0 + lc;
#pragma unroll
    for (int j = 0; j < 8; ++j) As[lr * 32 + lc + j] = f2bfbits(as[j]);
    const float* bs = Vh + (long long)(k0 + kr) * 1024 + n0 + kc;
#pragma unroll
    for (int j = 0; j < 8; ++j) Bs[kr * 64 + kc + j] = f2bfbits(bs[j]);
    __syncthreads();
    v16bf af = frag_a_lds(As, wr * 16, lane);
    v16bf b0 = frag_b_lds_n(Bs, wc * 32, lane);
    v16bf b1 = frag_b_lds_n(Bs, wc * 32 + 16, lane);
    acc0 = __builtin_amdgcn_wmma_f32_16x16x32_bf16(false, af, false, b0, (short)0, acc0, false, false);
    acc1 = __builtin_amdgcn_wmma_f32_16x16x32_bf16(false, af, false, b1, (short)0, acc1, false, false);
  }
  store_ctile(Oh, 1024, m0 + wr * 16, n0 + wc * 32,      lane, acc0, nullptr, 0);
  store_ctile(Oh, 1024, m0 + wr * 16, n0 + wc * 32 + 16, lane, acc1, nullptr, 0);
}

// ---------------- LayerNorm over D=1024 (+ fused epilogues) ------------------
// mode 0: y = LN(x)*g + b
// mode 1: y = gelu(LN(x)*g+b) * scalep[0]                       (modality proj)
// mode 2: y = sg*(LN(x)*g+b) + (1-sg)*extra, sg=sigmoid(scalep[0])  (attn gate)
__global__ __launch_bounds__(256) void layernorm_kernel(
    const float* __restrict__ X, const float* __restrict__ G, const float* __restrict__ Bt,
    float* __restrict__ Y, int mode, const float* __restrict__ scalep,
    const float* __restrict__ extra) {
  long long row = blockIdx.x;
  const float* x = X + row * 1024;
  float* y = Y + row * 1024;
  int tid = threadIdx.x;
  float v[4], s = 0.f, sq = 0.f;
#pragma unroll
  for (int j = 0; j < 4; ++j) {
    v[j] = x[tid + (j << 8)];
    s += v[j];
    sq += v[j] * v[j];
  }
  __shared__ float red[256];
  red[tid] = s; __syncthreads();
  for (int st = 128; st > 0; st >>= 1) { if (tid < st) red[tid] += red[tid + st]; __syncthreads(); }
  float mean = red[0] * (1.0f / 1024.0f);
  __syncthreads();
  red[tid] = sq; __syncthreads();
  for (int st = 128; st > 0; st >>= 1) { if (tid < st) red[tid] += red[tid + st]; __syncthreads(); }
  float var = red[0] * (1.0f / 1024.0f) - mean * mean;
  float rstd = rsqrtf(var + 1e-5f);
#pragma unroll
  for (int j = 0; j < 4; ++j) {
    int c = tid + (j << 8);
    float t = (v[j] - mean) * rstd * G[c] + Bt[c];
    if (mode == 1) {
      t = 0.5f * t * (1.0f + erff(t * 0.70710678118f));
      t *= scalep[0];
    } else if (mode == 2) {
      float sg = 1.0f / (1.0f + __expf(-scalep[0]));
      t = sg * t + (1.0f - sg) * extra[row * 1024 + c];
    }
    y[c] = t;
  }
}

// ---------------- Row softmax over 512 cols (scale fused) --------------------
__global__ __launch_bounds__(256) void softmax_kernel(float* __restrict__ X,
                                                      const float* __restrict__ scalep) {
  long long row = blockIdx.x;
  float* x = X + row * 512;
  int tid = threadIdx.x;
  float sc = scalep[0];
  float a = x[tid] * sc, b = x[tid + 256] * sc;
  __shared__ float red[256];
  red[tid] = fmaxf(a, b); __syncthreads();
  for (int st = 128; st > 0; st >>= 1) { if (tid < st) red[tid] = fmaxf(red[tid], red[tid + st]); __syncthreads(); }
  float mx = red[0];
  __syncthreads();
  float e0 = __expf(a - mx), e1 = __expf(b - mx);
  red[tid] = e0 + e1; __syncthreads();
  for (int st = 128; st > 0; st >>= 1) { if (tid < st) red[tid] += red[tid + st]; __syncthreads(); }
  float inv = 1.0f / red[0];
  x[tid] = e0 * inv;
  x[tid + 256] = e1 * inv;
}

// ---------------- Weight preprocessing: RPB ternary + residual quantile ------
// stats s[]: 0=sum|W| 1=sum kept|W| 2=kept count 3=max|R| 4=threshold
#define NBLK 240

__global__ void wclear_kernel(float* s, unsigned* hist) {
  int t = threadIdx.x;
  if (t < 16) s[t] = 0.0f;
  if (t < 512) hist[t] = 0u;
}

__global__ __launch_bounds__(256) void wabs_partial_kernel(const float* __restrict__ W,
                                                           long long n, float* __restrict__ partial) {
  __shared__ float red[256];
  int tid = threadIdx.x;
  float s = 0.f;
  for (long long i = (long long)blockIdx.x * 256 + tid; i < n; i += (long long)gridDim.x * 256)
    s += fabsf(W[i]);
  red[tid] = s; __syncthreads();
  for (int st = 128; st > 0; st >>= 1) { if (tid < st) red[tid] += red[tid + st]; __syncthreads(); }
  if (tid == 0) partial[blockIdx.x] = red[0];
}

__global__ void wsum1_kernel(const float* partial, int nb, float* s) {
  if (threadIdx.x == 0) { float t = 0.f; for (int i = 0; i < nb; ++i) t += partial[i]; s[0] = t; }
}

__global__ __launch_bounds__(256) void wkept_partial_kernel(const float* __restrict__ W,
                                                            long long n, const float* __restrict__ s,
                                                            float* __restrict__ partial) {
  float delta = 0.7f * s[0] / (float)n;
  __shared__ float red[256];
  int tid = threadIdx.x;
  float sa = 0.f, sc = 0.f;
  for (long long i = (long long)blockIdx.x * 256 + tid; i < n; i += (long long)gridDim.x * 256) {
    float a = fabsf(W[i]);
    if (a > delta) { sa += a; sc += 1.f; }
  }
  red[tid] = sa; __syncthreads();
  for (int st = 128; st > 0; st >>= 1) { if (tid < st) red[tid] += red[tid + st]; __syncthreads(); }
  if (tid == 0) partial[blockIdx.x] = red[0];
  __syncthreads();
  red[tid] = sc; __syncthreads();
  for (int st = 128; st > 0; st >>= 1) { if (tid < st) red[tid] += red[tid + st]; __syncthreads(); }
  if (tid == 0) partial[gridDim.x + blockIdx.x] = red[0];
}

__global__ void wsum2_kernel(const float* partial, int nb, float* s) {
  if (threadIdx.x == 0) {
    float a = 0.f, c = 0.f;
    for (int i = 0; i < nb; ++i) { a += partial[i]; c += partial[nb + i]; }
    s[1] = a; s[2] = c;
  }
}

__global__ __launch_bounds__(256) void wmaxr_kernel(const float* __restrict__ W,
                                                    long long n, float* __restrict__ s) {
  float delta = 0.7f * s[0] / (float)n;
  float alpha = s[1] / fmaxf(s[2], 1.0f);
  __shared__ float red[256];
  int tid = threadIdx.x;
  float mx = 0.f;
  for (long long i = (long long)blockIdx.x * 256 + tid; i < n; i += (long long)gridDim.x * 256) {
    float w = W[i], a = fabsf(w);
    float wq = (a > delta) ? (w > 0.f ? alpha : -alpha) : 0.f;
    mx = fmaxf(mx, fabsf(w - wq));
  }
  red[tid] = mx; __syncthreads();
  for (int st = 128; st > 0; st >>= 1) { if (tid < st) red[tid] = fmaxf(red[tid], red[tid + st]); __syncthreads(); }
  if (tid == 0) atomicMax((unsigned*)&s[3], __float_as_uint(red[0]));
}

__global__ __launch_bounds__(256) void whist_kernel(const float* __restrict__ W, long long n,
                                                    const float* __restrict__ s,
                                                    unsigned* __restrict__ hist) {
  float delta = 0.7f * s[0] / (float)n;
  float alpha = s[1] / fmaxf(s[2], 1.0f);
  float maxr = s[3];
  float inv = (maxr > 0.f) ? (512.0f / maxr) : 0.f;
  for (long long i = (long long)blockIdx.x * 256 + threadIdx.x; i < n;
       i += (long long)gridDim.x * 256) {
    float w = W[i], a = fabsf(w);
    float wq = (a > delta) ? (w > 0.f ? alpha : -alpha) : 0.f;
    int b = (int)(fabsf(w - wq) * inv);
    if (b > 511) b = 511;
    atomicAdd(&hist[b], 1u);
  }
}

__global__ void wthresh_kernel(float* s, const unsigned* hist, long long n, float ratio) {
  if (threadIdx.x == 0) {
    float maxr = s[3];
    long long target = (long long)((double)ratio * (double)n);
    if (target < 1) target = 1;
    long long cum = 0;
    float thr = 0.f;
    for (int b = 511; b >= 0; --b) {
      cum += hist[b];
      if (cum >= target) { thr = (float)b * maxr * (1.0f / 512.0f); break; }
    }
    s[4] = thr;
  }
}

__global__ __launch_bounds__(256) void wbuild_kernel(const float* __restrict__ W, long long n,
                                                     const float* __restrict__ s,
                                                     unsigned short* __restrict__ out) {
  float delta = 0.7f * s[0] / (float)n;
  float alpha = s[1] / fmaxf(s[2], 1.0f);
  float thr = s[4];
  long long i = (long long)blockIdx.x * 256 + threadIdx.x;
  if (i < n) {
    float w = W[i], a = fabsf(w);
    float wq = (a > delta) ? (w > 0.f ? alpha : -alpha) : 0.f;
    float r = w - wq;
    float we = wq + ((fabsf(r) >= thr) ? r : 0.f);
    out[i] = f2bfbits(we);
  }
}

// ---------------------------------------------------------------------------
// Host-side orchestration
// ---------------------------------------------------------------------------

static void prep_weight(const float* W, long long n, float ratio, unsigned short* out,
                        float* s, unsigned* hist, float* partial, hipStream_t st) {
  wclear_kernel<<<1, 512, 0, st>>>(s, hist);
  wabs_partial_kernel<<<NBLK, 256, 0, st>>>(W, n, partial);
  wsum1_kernel<<<1, 32, 0, st>>>(partial, NBLK, s);
  wkept_partial_kernel<<<NBLK, 256, 0, st>>>(W, n, s, partial);
  wsum2_kernel<<<1, 32, 0, st>>>(partial, NBLK, s);
  wmaxr_kernel<<<NBLK, 256, 0, st>>>(W, n, s);
  whist_kernel<<<NBLK, 256, 0, st>>>(W, n, s, hist);
  wthresh_kernel<<<1, 32, 0, st>>>(s, hist, n, ratio);
  wbuild_kernel<<<(int)((n + 255) / 256), 256, 0, st>>>(W, n, s, out);
}

struct AttnP {
  const float *lnq_g, *lnq_b, *lnk_g, *lnk_b, *lnv_g, *lnv_b, *lno_g, *lno_b;
  const float *bq, *bk, *bv, *bo, *gate, *scale;
  const unsigned short *wq, *wk, *wv, *wo;
};

static void run_attention(const AttnP& p, const float* query, const float* key, const float* value,
                          float* qn, float* kn, float* vn, float* q, float* k, float* v,
                          float* o, float* scores, float* tmp, float* out, hipStream_t st) {
  layernorm_kernel<<<4096, 256, 0, st>>>(query, p.lnq_g, p.lnq_b, qn, 0, nullptr, nullptr);
  layernorm_kernel<<<4096, 256, 0, st>>>(key,   p.lnk_g, p.lnk_b, kn, 0, nullptr, nullptr);
  layernorm_kernel<<<4096, 256, 0, st>>>(value, p.lnv_g, p.lnv_b, vn, 0, nullptr, nullptr);
  gemm_bias_kernel<<<dim3(16, 64), 256, 0, st>>>(qn, 1024, p.wq, p.bq, q, 1024, 1024, 1);
  gemm_bias_kernel<<<dim3(16, 64), 256, 0, st>>>(kn, 1024, p.wk, p.bk, k, 1024, 1024, 1);
  gemm_bias_kernel<<<dim3(16, 64), 256, 0, st>>>(vn, 1024, p.wv, p.bv, v, 1024, 1024, 1);
  for (int b = 0; b < 8; ++b) {
    long long off = (long long)b * 512 * 1024;
    attn_scores_kernel<<<dim3(8, 8, 8), 256, 0, st>>>(q + off, k + off, scores);
    softmax_kernel<<<8 * 512, 256, 0, st>>>(scores, p.scale);
    attn_out_kernel<<<dim3(2, 8, 8), 256, 0, st>>>(scores, v + off, o + off);
  }
  gemm_bias_kernel<<<dim3(16, 64), 256, 0, st>>>(o, 1024, p.wo, p.bo, tmp, 1024, 1024, 0);
  layernorm_kernel<<<4096, 256, 0, st>>>(tmp, p.lno_g, p.lno_b, out, 2, p.gate, qn);
}

extern "C" void kernel_launch(void* const* d_in, const int* in_sizes, int n_in,
                              void* d_out, int out_size, void* d_ws, size_t ws_size,
                              hipStream_t stream) {
  (void)in_sizes; (void)n_in; (void)out_size; (void)ws_size;
  auto F = [&](int i) { return (const float*)d_in[i]; };

  // Input flattening order (setup_inputs dict order, depth-first):
  // 0 text, 1 image,
  // 2..5 proj_text {W,b,ln_g,ln_b}, 6..9 proj_image {W,b,ln_g,ln_b},
  // 10..27 t2i {Wq,bq,Wk,bk,Wv,bv,Wo,bo, lnq_g,lnq_b,lnk_g,lnk_b,lnv_g,lnv_b,lno_g,lno_b, gate,scale},
  // 28..45 i2t (same layout),
  // 46,47 align_text {W,b}, 48,49 align_image {W,b}, 50,51 final {W,b},
  // 52 ln_g, 53 ln_b, 54 scale_text, 55 scale_image
  const float* text  = F(0);
  const float* image = F(1);

  // -------- workspace bump allocator --------
  char* ws = (char*)d_ws;
  size_t off = 0;
  auto alloc = [&](size_t bytes) -> char* {
    char* p = ws + off;
    off += (bytes + 255) & ~(size_t)255;
    return p;
  };

  // bf16 preprocessed weights
  const int      widx[13]  = {2, 6, 10, 12, 14, 16, 28, 30, 32, 34, 46, 48, 50};
  const long long wn[13]   = {1024LL * 768, 1048576, 1048576, 1048576, 1048576, 1048576,
                              1048576, 1048576, 1048576, 1048576, 1048576, 1048576, 2097152};
  const float    wrat[13]  = {0.15f, 0.15f, 0.15f, 0.15f, 0.15f, 0.2f,
                              0.15f, 0.15f, 0.15f, 0.2f, 0.2f, 0.2f, 0.2f};
  unsigned short* wbf[13];
  for (int i = 0; i < 13; ++i) wbf[i] = (unsigned short*)alloc((size_t)wn[i] * 2);

  float*    stats   = (float*)alloc(64);
  unsigned* hist    = (unsigned*)alloc(512 * 4);
  float*    partial = (float*)alloc(2 * NBLK * 4);

  const size_t ABUF = (size_t)4096 * 1024 * 4;   // 16 MB activation buffer
  float* buf_t   = (float*)alloc(ABUF);
  float* buf_i   = (float*)alloc(ABUF);
  float* buf_t2i = (float*)alloc(ABUF);
  float* buf_i2t = (float*)alloc(ABUF);
  float* buf_qn  = (float*)alloc(ABUF);
  float* buf_tmp = (float*)alloc(ABUF);          // pre-LN scratch (kn-free phase)
  float* buf_kn  = (float*)alloc(ABUF);
  float* buf_vn  = (float*)alloc(ABUF);
  float* buf_q   = (float*)alloc(ABUF);          // cat rows 0..2047 alias here later
  float* buf_k   = (float*)alloc(ABUF);          // cat rows 2048..4095 alias here (contiguous)
  float* buf_v   = (float*)alloc(ABUF);
  float* buf_o   = (float*)alloc(ABUF);
  float* scores  = (float*)alloc((size_t)8 * 512 * 512 * 4);   // 8 MB, per-batch reuse

  // -------- 1) preprocess all 13 RPB weight matrices to bf16 --------
  for (int i = 0; i < 13; ++i)
    prep_weight(F(widx[i]), wn[i], wrat[i], wbf[i], stats, hist, partial, stream);

  // -------- 2) modality projections --------
  gemm_bias_kernel<<<dim3(16, 64), 256, 0, stream>>>(text, 768, wbf[0], F(3), buf_tmp, 1024, 768, 0);
  layernorm_kernel<<<4096, 256, 0, stream>>>(buf_tmp, F(4), F(5), buf_t, 1, F(54), nullptr);
  gemm_bias_kernel<<<dim3(16, 64), 256, 0, stream>>>(image, 1024, wbf[1], F(7), buf_tmp, 1024, 1024, 0);
  layernorm_kernel<<<4096, 256, 0, stream>>>(buf_tmp, F(8), F(9), buf_i, 1, F(55), nullptr);

  // -------- 3) cross attentions --------
  auto attn_params = [&](int base, int wb) {
    AttnP p;
    p.bq = F(base + 1); p.bk = F(base + 3); p.bv = F(base + 5); p.bo = F(base + 7);
    p.lnq_g = F(base + 8);  p.lnq_b = F(base + 9);
    p.lnk_g = F(base + 10); p.lnk_b = F(base + 11);
    p.lnv_g = F(base + 12); p.lnv_b = F(base + 13);
    p.lno_g = F(base + 14); p.lno_b = F(base + 15);
    p.gate  = F(base + 16); p.scale = F(base + 17);
    p.wq = wbf[wb]; p.wk = wbf[wb + 1]; p.wv = wbf[wb + 2]; p.wo = wbf[wb + 3];
    return p;
  };
  AttnP pt2i = attn_params(10, 2);
  AttnP pi2t = attn_params(28, 6);
  run_attention(pt2i, buf_t, buf_i, buf_i, buf_qn, buf_kn, buf_vn,
                buf_q, buf_k, buf_v, buf_o, scores, buf_tmp, buf_t2i, stream);
  run_attention(pi2t, buf_i, buf_t, buf_t, buf_qn, buf_kn, buf_vn,
                buf_q, buf_k, buf_v, buf_o, scores, buf_tmp, buf_i2t, stream);

  // -------- 4) align + concat + final + layernorm --------
  float* cat = buf_q;  // 32 MB spanning buf_q..buf_k (contiguous, both now dead)
  gemm_bias_kernel<<<dim3(16, 64), 256, 0, stream>>>(buf_t2i, 1024, wbf[10], F(47), cat, 2048, 1024, 0);
  gemm_bias_kernel<<<dim3(16, 64), 256, 0, stream>>>(buf_i2t, 1024, wbf[11], F(49), cat + 1024, 2048, 1024, 0);
  gemm_bias_kernel<<<dim3(16, 64), 256, 0, stream>>>(cat, 2048, wbf[12], F(51), buf_tmp, 1024, 2048, 0);
  layernorm_kernel<<<4096, 256, 0, stream>>>(buf_tmp, F(52), F(53), (float*)d_out, 0, nullptr, nullptr);
}
